// TabCellEmb_42717744726717
// MI455X (gfx1250) — matrix operand order repaired and checked by reading, not compiled, not run on hardware
//
#include <hip/hip_runtime.h>
#include <hip/hip_bf16.h>
#include <math.h>

// ---------------------------------------------------------------------------
// TabCellEmb for MI455X (gfx1250, wave32, WMMA).
//
// Pipeline:
//   K1 fgate_kernel : fuse-MLP evaluated for the 8 DISTINCT types only
//   K2 cne_kernel   : masked cn-token average + dt_e*fgate (float4 path)
//   K3 packA_kernel : cn_e -> bf16 in WMMA A-fragment layout, k = 16*half + i
//   K4 packW_kernel : gate_w1 -> bf16 in WMMA B-fragment order
//   K5 gate_gemm    : v_wmma_f32_16x16x32_bf16 GEMM (M=4096,K=768,N=3072)
//                     fused relu/+b1/*w2 reduction + sigmoid -> g[4096];
//                     A staged in LDS, anti-LICM barrier keeps ds_loads
//                     in-loop (prevents the scratch-spill seen previously)
//   K6 scan_kernel  : wave32 ballot prefix scan of cv_mask
//   K7 cls_kernel   : CLS rows (float4)
//   K8 zero_cv      : exact-zero fill of ragged cv region (float4)
//   K9 scatter      : out[row, 65+pos, :] = word_emb[cv_ids] + g*cn_e (float4)
// ---------------------------------------------------------------------------

typedef __attribute__((ext_vector_type(16))) __bf16 v16bf;
typedef __attribute__((ext_vector_type(8)))  float  v8f;
typedef __bf16 bf16_t;

#define ROWS   64
#define CELLS  64
#define NCELL  4096
#define CN_LEN 8
#define CV_LEN 16
#define EMB    768
#define EMB4   192             // EMB / 4
#define HID    3072
#define L_CV   1024
#define TOKENS 1089            // 1 + CELLS + L_CV
#define KT     24              // EMB / 32
#define NT     192             // HID / 16
#define MT     256             // NCELL / 16

__device__ __forceinline__ float sigmoidf_(float x) {
    return 1.f / (1.f + __expf(-x));
}

// --- K1: fuse MLP for the 8 distinct datatype embeddings ------------------
__global__ void fgate_kernel(const float* __restrict__ type_emb,
                             const float* __restrict__ fuse_w1,
                             const float* __restrict__ fuse_b1,
                             const float* __restrict__ fuse_w2,
                             const float* __restrict__ fuse_b2,
                             float* __restrict__ fgate) {
    int t = blockIdx.x;                         // 0..7
    __shared__ float te[EMB];
    for (int i = threadIdx.x; i < EMB; i += blockDim.x) te[i] = type_emb[t * EMB + i];
    __syncthreads();
    float partial = 0.f;
    for (int h = threadIdx.x; h < HID; h += blockDim.x) {
        float d = fuse_b1[h];
        for (int k = 0; k < EMB; ++k) d += te[k] * fuse_w1[(size_t)k * HID + h];
        d = d > 0.f ? d : 0.f;
        partial += d * fuse_w2[h];
    }
    __shared__ float red[256];
    red[threadIdx.x] = partial;
    __syncthreads();
    for (int s = blockDim.x >> 1; s > 0; s >>= 1) {
        if (threadIdx.x < (unsigned)s) red[threadIdx.x] += red[threadIdx.x + s];
        __syncthreads();
    }
    if (threadIdx.x == 0) fgate[t] = sigmoidf_(red[0] + fuse_b2[0]);
}

// --- K2: cn_e = masked avg of column-name tokens + dt_e * fgate (float4) --
__global__ void cne_kernel(const int* __restrict__ cn_ids,
                           const int* __restrict__ cn_mask,
                           const int* __restrict__ c_types,
                           const float* __restrict__ word_emb,
                           const float* __restrict__ type_emb,
                           const float* __restrict__ fgate,
                           float* __restrict__ cn_e,
                           float* __restrict__ out) {
    int cell = blockIdx.x;                      // 0..4095
    int row  = cell >> 6, col = cell & 63;
    int id[CN_LEN], mk[CN_LEN];
    int cnt = 0;
    #pragma unroll
    for (int j = 0; j < CN_LEN; ++j) {
        id[j] = cn_ids[cell * CN_LEN + j];
        mk[j] = cn_mask[cell * CN_LEN + j];
        cnt  += mk[j];
    }
    float inv = 1.f / (float)cnt;               // >=1 guaranteed by setup
    int   ty  = c_types[cell];
    float fg  = fgate[ty];
    const float4* dt4 = (const float4*)(type_emb + (size_t)ty * EMB);
    float4*       cn4 = (float4*)(cn_e + (size_t)cell * EMB);
    float4*       ot4 = (float4*)(out + ((size_t)row * TOKENS + 1 + col) * EMB);

    int e4 = threadIdx.x;                       // blockDim = 192
    float4 s = make_float4(0.f, 0.f, 0.f, 0.f);
    #pragma unroll
    for (int j = 0; j < CN_LEN; ++j) {
        if (mk[j]) {
            float4 w = ((const float4*)(word_emb + (size_t)id[j] * EMB))[e4];
            s.x += w.x; s.y += w.y; s.z += w.z; s.w += w.w;
        }
    }
    float4 d = dt4[e4];
    float4 v;
    v.x = s.x * inv + d.x * fg;
    v.y = s.y * inv + d.y * fg;
    v.z = s.z * inv + d.z * fg;
    v.w = s.w * inv + d.w * fg;
    cn4[e4] = v;
    ot4[e4] = v;
}

// --- K3: pack A (cn_e) into WMMA 16-bit A 16x32 fragment order ------------
__global__ void packA_kernel(const float* __restrict__ cn_e,
                             bf16_t* __restrict__ Apk) {
    int idx = blockIdx.x * blockDim.x + threadIdx.x;   // < 4096*768
    if (idx >= NCELL * EMB) return;
    int i    = idx & 15;
    int lane = (idx >> 4) & 31;
    int kt   = (idx >> 9) % KT;
    int mt   = idx / (KT * 512);
    int m = lane & 15, half = lane >> 4;
    int group = i >> 3, within = i & 7;
    int k = kt * 32 + 16 * half + i;
    Apk[idx] = (bf16_t)cn_e[(size_t)(mt * 16 + m) * EMB + k];
}

// --- K4: pack gate_w1 into WMMA 16-bit B 32x16 fragment order -------------
__global__ void packW_kernel(const float* __restrict__ gate_w1,
                             bf16_t* __restrict__ Wpk) {
    int idx = blockIdx.x * blockDim.x + threadIdx.x;   // < 3072*768
    if (idx >= HID * EMB) return;
    int i    = idx & 15;
    int lane = (idx >> 4) & 31;
    int kt   = (idx >> 9) % KT;
    int nt   = idx / (KT * 512);
    int k = kt * 32 + 16 * (lane >> 4) + i;
    int n = nt * 16 + (lane & 15);
    Wpk[idx] = (bf16_t)gate_w1[(size_t)k * HID + n];
}

// --- K5: WMMA GEMM + fused relu/b1/w2 reduction + sigmoid -> g[4096] ------
__global__ void gate_gemm_kernel(const bf16_t* __restrict__ Apk,
                                 const bf16_t* __restrict__ Wpk,
                                 const float* __restrict__ gate_b1,
                                 const float* __restrict__ gate_w2,
                                 const float* __restrict__ gate_b2,
                                 float* __restrict__ g) {
    int mt   = blockIdx.x;                      // 0..255 : 16-cell M tile
    int wave = threadIdx.x >> 5;                // 8 waves
    int lane = threadIdx.x & 31;
    int half = lane >> 4;
    int n16  = lane & 15;

    // Stage this M-tile's packed A (24 K-tiles x 32 lanes x 16 bf16 = 24KB)
    __shared__ __align__(32) bf16_t Alds[KT * 32 * 16];
    {
        const uint4* Ag = (const uint4*)(Apk + (size_t)mt * KT * 512);
        uint4* Al = (uint4*)Alds;
        for (int i = threadIdx.x; i < KT * 512 * 2 / 16; i += blockDim.x) Al[i] = Ag[i];
    }
    __syncthreads();

    const v16bf* Al16 = (const v16bf*)Alds;
    float accs[8] = {0.f, 0.f, 0.f, 0.f, 0.f, 0.f, 0.f, 0.f};

    for (int nt = wave; nt < NT; nt += 8) {
        v8f c = {};
        const v16bf* Bp = (const v16bf*)Wpk + (size_t)nt * KT * 32;
        for (int kt = 0; kt < KT; ++kt) {
            // Anti-LICM barrier: without it the compiler hoists all 24 A
            // fragments out of the nt loop (192 VGPRs) and spills them to
            // scratch; with it, each fragment is a fresh ds_load_b128 pair
            // feeding the WMMA directly from LDS.
            asm volatile("" ::: "memory");
            v16bf a = Al16[kt * 32 + lane];
            v16bf b = Bp[kt * 32 + lane];
            c = __builtin_amdgcn_wmma_f32_16x16x32_bf16(
                    false, a, false, b, (short)0, c, false, false);
        }
        int   n  = nt * 16 + n16;
        float b1 = gate_b1[n];
        float w2 = gate_w2[n];
        #pragma unroll
        for (int j = 0; j < 8; ++j) {           // c[j] = C[M = 8*half+j][n]
            float t = c[j] + b1;
            t = t > 0.f ? t : 0.f;
            accs[j] += t * w2;
        }
    }
    // reduce over the 16 lanes (N dimension) inside each half
    #pragma unroll
    for (int j = 0; j < 8; ++j) {
        float v = accs[j];
        v += __shfl_xor(v, 1, 32);
        v += __shfl_xor(v, 2, 32);
        v += __shfl_xor(v, 4, 32);
        v += __shfl_xor(v, 8, 32);
        accs[j] = v;
    }
    __shared__ float msum[16];
    if (threadIdx.x < 16) msum[threadIdx.x] = 0.f;
    __syncthreads();
    if (n16 == 0) {
        #pragma unroll
        for (int j = 0; j < 8; ++j) atomicAdd(&msum[half * 8 + j], accs[j]);
    }
    __syncthreads();
    if (threadIdx.x < 16)
        g[mt * 16 + threadIdx.x] = sigmoidf_(msum[threadIdx.x] + gate_b2[0]);
}

// --- K6: per-row ragged prefix scan of cv_mask (wave32 ballot) ------------
__global__ void scan_kernel(const unsigned char* __restrict__ cv_mask,
                            int* __restrict__ pos) {
    int row  = blockIdx.x;                      // 64 rows, blockDim = 32
    int lane = threadIdx.x;
    int base = row * L_CV;
    int running = 0;
    for (int c0 = 0; c0 < L_CV; c0 += 32) {
        int m = cv_mask[base + c0 + lane] ? 1 : 0;
        unsigned int bb = (unsigned int)__ballot(m);
        int below = __popc(bb & ((1u << lane) - 1u));
        pos[base + c0 + lane] = m ? (running + below) : -1;
        running += __popc(bb);
    }
}

// --- K7: CLS token (float4) ----------------------------------------------
__global__ void cls_kernel(const float* __restrict__ cls_w,
                           float* __restrict__ out) {
    int idx = blockIdx.x * blockDim.x + threadIdx.x;   // < 64*192
    int row = idx / EMB4, e4 = idx % EMB4;
    ((float4*)(out + (size_t)row * TOKENS * EMB))[e4] =
        ((const float4*)cls_w)[e4];
}

// --- K8: exact-zero fill of the cv region (pad_zeros semantics) -----------
__global__ void zero_cv_kernel(float* __restrict__ out) {
    int row = blockIdx.y;
    size_t off = ((size_t)row * TOKENS + 1 + CELLS) * EMB;
    size_t i = (size_t)(blockIdx.x * blockDim.x + threadIdx.x) * 4;
    float4 z = make_float4(0.f, 0.f, 0.f, 0.f);
    *(float4*)(out + off + i) = z;
}

// --- K9: scatter linked values into compacted positions (float4) ----------
__global__ void scatter_kernel(const int* __restrict__ cv_ids,
                               const int* __restrict__ pos,
                               const float* __restrict__ cn_e,
                               const float* __restrict__ g,
                               const float* __restrict__ word_emb,
                               float* __restrict__ out) {
    int tok = blockIdx.x;                       // 0..65535, = row*1024 + local
    int p = pos[tok];
    if (p < 0) return;
    int cell = tok >> 4;
    int row  = cell >> 6;
    int wid  = cv_ids[tok];
    float gg = g[cell];
    const float4* we4 = (const float4*)(word_emb + (size_t)wid * EMB);
    const float4* ce4 = (const float4*)(cn_e + (size_t)cell * EMB);
    float4* ot4 = (float4*)(out + ((size_t)row * TOKENS + 1 + CELLS + p) * EMB);
    int e4 = threadIdx.x;                       // blockDim = 192
    float4 w = we4[e4];
    float4 c = ce4[e4];
    float4 v;
    v.x = w.x + c.x * gg;
    v.y = w.y + c.y * gg;
    v.z = w.z + c.z * gg;
    v.w = w.w + c.w * gg;
    ot4[e4] = v;
}

// ---------------------------------------------------------------------------
extern "C" void kernel_launch(void* const* d_in, const int* in_sizes, int n_in,
                              void* d_out, int out_size, void* d_ws, size_t ws_size,
                              hipStream_t stream) {
    (void)in_sizes; (void)n_in; (void)out_size; (void)ws_size;
    const int*   cn_ids   = (const int*)d_in[0];
    const int*   cn_mask  = (const int*)d_in[1];
    const int*   c_types  = (const int*)d_in[2];
    const int*   cv_ids   = (const int*)d_in[3];
    const unsigned char* cv_mask = (const unsigned char*)d_in[4];  // jnp bool = 1 byte
    const float* word_emb = (const float*)d_in[5];
    const float* type_emb = (const float*)d_in[6];
    const float* fuse_w1  = (const float*)d_in[7];
    const float* fuse_b1  = (const float*)d_in[8];
    const float* fuse_w2  = (const float*)d_in[9];
    const float* fuse_b2  = (const float*)d_in[10];
    const float* gate_w1  = (const float*)d_in[11];
    const float* gate_b1  = (const float*)d_in[12];
    const float* gate_w2  = (const float*)d_in[13];
    const float* gate_b2  = (const float*)d_in[14];
    const float* cls_w    = (const float*)d_in[15];
    float* out = (float*)d_out;

    // workspace layout (all regions 256B-aligned; total ~23.9 MB)
    char* ws = (char*)d_ws;
    float*  fgate = (float*)(ws);                                   //     256 B
    float*  cn_e  = (float*)(ws + 256);                             // 12.58 MB
    bf16_t* Apk   = (bf16_t*)(ws + 256 + 12582912);                 //  6.29 MB
    bf16_t* Wpk   = (bf16_t*)(ws + 256 + 12582912 + 6291456);       //  4.72 MB
    float*  g     = (float*)(ws + 256 + 12582912 + 6291456 + 4718592);
    int*    pos   = (int*)(ws + 256 + 12582912 + 6291456 + 4718592 + 16384);

    fgate_kernel<<<8, 256, 0, stream>>>(type_emb, fuse_w1, fuse_b1, fuse_w2,
                                        fuse_b2, fgate);
    cne_kernel<<<NCELL, EMB4, 0, stream>>>(cn_ids, cn_mask, c_types, word_emb,
                                           type_emb, fgate, cn_e, out);
    packA_kernel<<<(NCELL * EMB + 255) / 256, 256, 0, stream>>>(cn_e, Apk);
    packW_kernel<<<(HID * EMB + 255) / 256, 256, 0, stream>>>(gate_w1, Wpk);
    gate_gemm_kernel<<<MT, 256, 0, stream>>>(Apk, Wpk, gate_b1, gate_w2,
                                             gate_b2, g);
    scan_kernel<<<ROWS, 32, 0, stream>>>(cv_mask, pos);
    cls_kernel<<<(ROWS * EMB4) / 256, 256, 0, stream>>>(cls_w, out);
    {
        dim3 zgrid(L_CV * EMB / 4 / 256, ROWS);   // 768 x 64 blocks of float4 stores
        zero_cv_kernel<<<zgrid, 256, 0, stream>>>(out);
    }
    scatter_kernel<<<NCELL * CV_LEN, EMB4, 0, stream>>>(cv_ids, pos, cn_e, g,
                                                        word_emb, out);
}
